// SoftQuantizer_57578331570649
// MI455X (gfx1250) — compile-verified, matrix operand order checked by reading, not run on hardware
//
#include <hip/hip_runtime.h>

typedef __attribute__((ext_vector_type(16))) _Float16     v16h;
typedef __attribute__((ext_vector_type(8)))  float        v8f;
typedef __attribute__((ext_vector_type(4)))  float        v4f;
typedef __attribute__((ext_vector_type(4)))  unsigned int v4u;
typedef __attribute__((ext_vector_type(8)))  int          v8i;
typedef __attribute__((ext_vector_type(4)))  int          v4i;

#define NUM_BINS 256
#define EMBED    256
#define ROWS_PER_BLOCK 128   // 8 waves x 16 rows
#define COLS_PER_BLOCK 128   // 8 N-tiles of 16

// ---------------------------------------------------------------------------
// Prep kernel (runs every call, tiny): builds
//   negS[k] = -exp(log_scale[k]) * log2(e)           (256 floats)
//   frag[]  = codebook as f16 WMMA B-fragments       (128 frags x 1KB = 128KB)
// Fragment f = nt_global*8 + c holds half slot [L][h]:
//   n = nt_global*16 + (L&15),  k = c*32 + (L>=16?8:0) + (h<8?h:h+8)
// ---------------------------------------------------------------------------
__global__ __launch_bounds__(256)
void softq_prep(const float* __restrict__ LS, const float* __restrict__ CB,
                float* __restrict__ negS, _Float16* __restrict__ frag)
{
    const int t  = threadIdx.x;   // n = 0..255
    const int bx = blockIdx.x;    // 32 blocks, 8 k-rows each
    const float LOG2E = 1.4426950408889634f;
    if (bx == 0)
        negS[t] = -__builtin_amdgcn_exp2f(LS[t] * LOG2E) * LOG2E;

    const int n  = t;
    const int nt = n >> 4;
    #pragma unroll
    for (int i = 0; i < 8; ++i) {
        const int k  = bx * 8 + i;
        const int c  = k >> 5;
        const int kl = k & 31;
        const int hi = (kl >> 3) & 1;
        const int h  = (kl & 7) + ((kl >= 16) ? 8 : 0);
        const int L  = hi * 16 + (n & 15);
        const int f  = nt * 8 + c;
        frag[f * 512 + L * 16 + h] = (_Float16)CB[k * EMBED + n];
    }
}

// ---------------------------------------------------------------------------
// Main kernel (ws path): TDM copies this block's 64KB fragment half ws->LDS
// while lanes compute softmax numerators straight into A fragments.
// ---------------------------------------------------------------------------
__global__ __launch_bounds__(256)
void softq_main(const float* __restrict__ X, const float* __restrict__ Q,
                const float* __restrict__ negS, const _Float16* __restrict__ frag,
                float* __restrict__ emb_out, float* __restrict__ probs_out)
{
    __shared__ _Float16 cbLDS[64 * 512]; // 64KB

    const int t    = threadIdx.x;
    const int wv   = t >> 5;
    const int lane = t & 31;
    const int r    = lane & 15;
    const bool hi  = lane >= 16;
    const int bx   = blockIdx.x;
    const int by   = blockIdx.y;

    // ---- wave 0 kicks off async TDM copy: 64KB linear, data_size=8B ----
    if (wv == 0) {
        const unsigned long long ga =
            (unsigned long long)(const void*)(frag + (size_t)by * 64 * 512);
        const unsigned int lds_base =
            (unsigned int)(unsigned long long)(const void*)&cbLDS[0];
        const v4u g0 = { 1u,                                    // count=1
                         lds_base,                              // lds_addr
                         (unsigned int)(ga & 0xffffffffu),      // global_addr lo
                         (unsigned int)((ga >> 32) & 0x01ffffffu) | 0x80000000u }; // type=2
        const v8i g1 = { (int)0x00030000,     // wg_mask=0, data_size=3 (8B)
                         (int)(8192u << 16),  // tensor_dim0 lo16 (8192) @bit48
                         (int)(1u << 16),     // tensor_dim0 hi=0 | tensor_dim1 lo16=1
                         (int)(8192u << 16),  // tensor_dim1 hi=0 | tile_dim0=8192
                         (int)0x00010001,     // tile_dim1=1 | tile_dim2=1
                         (int)8192,           // tensor_dim0_stride lo32
                         0, 0 };
        const v4i g2 = { 1,                   // tensor_dim2=1
                         1,                   // tensor_dim3=1
                         8192,                // tensor_dim2_stride lo32
                         (int)(1u << 16) };   // stride hi=0 | tile_dim3=1
        const v4i g3 = { 8192,                // tensor_dim3_stride lo32
                         (int)(1u << 16),     // stride hi=0 | tensor_dim4 lo16=1
                         (int)(1u << 16),     // dim4 hi=0 | tile_dim4=1
                         0 };
        const v8i g4 = { 0, 0, 0, 0, 0, 0, 0, 0 };
        __builtin_amdgcn_tensor_load_to_lds(g0, g1, g2, g3, g4, 0);
    }

    // ---- softmax numerators straight into A fragments (overlaps TDM) ----
    const int   m      = bx * ROWS_PER_BLOCK + wv * 16 + r;
    const float xv     = X[m];
    const int   kbase0 = hi ? 8 : 0;

    v16h afrag[8];
    float sum = 0.0f;
    #pragma unroll
    for (int c = 0; c < 8; ++c) {
        #pragma unroll
        for (int g = 0; g < 4; ++g) {
            const int kstart = c * 32 + kbase0 + (g < 2 ? 4 * g : 4 * g + 8);
            const v4f q4 = *(const v4f*)(Q    + kstart);
            const v4f s4 = *(const v4f*)(negS + kstart);
            #pragma unroll
            for (int j = 0; j < 4; ++j) {
                const float e =
                    __builtin_amdgcn_exp2f(s4[j] * __builtin_fabsf(xv - q4[j]));
                sum += e;
                afrag[c][g * 4 + j] = (_Float16)e;
            }
        }
    }
    const float total = sum + __shfl_xor(sum, 16, 32);
    const float inv   = 1.0f / total;

    // ---- probs output (only the by==0 half of the grid) ----
    if (by == 0) {
        float* pr = probs_out + (size_t)m * NUM_BINS;
        #pragma unroll
        for (int c = 0; c < 8; ++c) {
            #pragma unroll
            for (int g = 0; g < 4; ++g) {
                const int kstart = c * 32 + kbase0 + (g < 2 ? 4 * g : 4 * g + 8);
                v4f p;
                #pragma unroll
                for (int j = 0; j < 4; ++j)
                    p[j] = (float)afrag[c][g * 4 + j] * inv;
                *(v4f*)(pr + kstart) = p;
            }
        }
    }

    float rs[8];
    #pragma unroll
    for (int i = 0; i < 8; ++i)
        rs[i] = __shfl(inv, (hi ? 8 : 0) + i, 32);

    if (wv == 0) __builtin_amdgcn_s_wait_tensorcnt(0);
    __syncthreads();

    // ---- GEMM: whole B tile loaded as one ds clause, then 8 WMMAs ----
    const int mrow_base = bx * ROWS_PER_BLOCK + wv * 16 + (hi ? 8 : 0);
    const int ncol_base = by * COLS_PER_BLOCK;
    #pragma unroll 1
    for (int nt = 0; nt < 8; ++nt) {
        const _Float16* bp = cbLDS + (nt * 8) * 512 + lane * 16;
        v16h b[8];
        #pragma unroll
        for (int c = 0; c < 8; ++c)
            b[c] = *(const v16h*)(bp + c * 512);
        __builtin_amdgcn_sched_barrier(0);   // keep all 16 ds_loads ahead of WMMAs
        v8f acc = {};
        #pragma unroll
        for (int c = 0; c < 8; ++c)
            acc = __builtin_amdgcn_wmma_f32_16x16x32_f16(
                      false, afrag[c], false, b[c], (short)0, acc, false, false);
        const int col = ncol_base + nt * 16 + r;
        #pragma unroll
        for (int i = 0; i < 8; ++i)
            emb_out[(size_t)(mrow_base + i) * EMBED + col] = acc[i] * rs[i];
    }
}

// ---------------------------------------------------------------------------
// Fallback (no workspace): fused kernel, per-block LDS fragmentation.
// ---------------------------------------------------------------------------
__global__ __launch_bounds__(256)
void softq_fused(const float* __restrict__ X, const float* __restrict__ Q,
                 const float* __restrict__ LS, const float* __restrict__ CB,
                 float* __restrict__ emb_out, float* __restrict__ probs_out)
{
    __shared__ _Float16 cbLDS[64 * 512];

    const int t    = threadIdx.x;
    const int wv   = t >> 5;
    const int lane = t & 31;
    const int r    = lane & 15;
    const bool hi  = lane >= 16;
    const int bx   = blockIdx.x;
    const int by   = blockIdx.y;
    const int ncol_base = by * COLS_PER_BLOCK;

    for (int idx = t; idx < 64 * 512; idx += 256) {
        const int f   = idx >> 9;
        const int rem = idx & 511;
        const int L   = rem >> 4;
        const int h   = rem & 15;
        const int nt  = f >> 3;
        const int c   = f & 7;
        const int n   = ncol_base + nt * 16 + (L & 15);
        const int k   = c * 32 + ((L >= 16) ? 8 : 0) + (h < 8 ? h : h + 8);
        cbLDS[idx] = (_Float16)CB[k * EMBED + n];
    }

    const int   m      = bx * ROWS_PER_BLOCK + wv * 16 + r;
    const float xv     = X[m];
    const int   kbase0 = hi ? 8 : 0;
    const float LOG2E  = 1.4426950408889634f;

    v16h afrag[8];
    float sum = 0.0f;
    #pragma unroll
    for (int c = 0; c < 8; ++c) {
        #pragma unroll
        for (int g = 0; g < 4; ++g) {
            const int kstart = c * 32 + kbase0 + (g < 2 ? 4 * g : 4 * g + 8);
            const v4f q4 = *(const v4f*)(Q  + kstart);
            const v4f l4 = *(const v4f*)(LS + kstart);
            #pragma unroll
            for (int j = 0; j < 4; ++j) {
                const float s = __builtin_amdgcn_exp2f(l4[j] * LOG2E);
                const float e = __builtin_amdgcn_exp2f(-s * LOG2E *
                                    __builtin_fabsf(xv - q4[j]));
                sum += e;
                afrag[c][g * 4 + j] = (_Float16)e;
            }
        }
    }
    const float total = sum + __shfl_xor(sum, 16, 32);
    const float inv   = 1.0f / total;

    if (by == 0) {
        float* pr = probs_out + (size_t)m * NUM_BINS;
        #pragma unroll
        for (int c = 0; c < 8; ++c) {
            #pragma unroll
            for (int g = 0; g < 4; ++g) {
                const int kstart = c * 32 + kbase0 + (g < 2 ? 4 * g : 4 * g + 8);
                v4f p;
                #pragma unroll
                for (int j = 0; j < 4; ++j)
                    p[j] = (float)afrag[c][g * 4 + j] * inv;
                *(v4f*)(pr + kstart) = p;
            }
        }
    }

    float rs[8];
    #pragma unroll
    for (int i = 0; i < 8; ++i)
        rs[i] = __shfl(inv, (hi ? 8 : 0) + i, 32);

    __syncthreads();

    const int mrow_base = bx * ROWS_PER_BLOCK + wv * 16 + (hi ? 8 : 0);
    #pragma unroll 1
    for (int nt = 0; nt < 8; ++nt) {
        const _Float16* bp = cbLDS + (nt * 8) * 512 + lane * 16;
        v16h b[8];
        #pragma unroll
        for (int c = 0; c < 8; ++c)
            b[c] = *(const v16h*)(bp + c * 512);
        __builtin_amdgcn_sched_barrier(0);
        v8f acc = {};
        #pragma unroll
        for (int c = 0; c < 8; ++c)
            acc = __builtin_amdgcn_wmma_f32_16x16x32_f16(
                      false, afrag[c], false, b[c], (short)0, acc, false, false);
        const int col = ncol_base + nt * 16 + r;
        #pragma unroll
        for (int i = 0; i < 8; ++i)
            emb_out[(size_t)(mrow_base + i) * EMBED + col] = acc[i] * rs[i];
    }
}

extern "C" void kernel_launch(void* const* d_in, const int* in_sizes, int n_in,
                              void* d_out, int out_size, void* d_ws, size_t ws_size,
                              hipStream_t stream) {
    const float* X  = (const float*)d_in[0];
    const float* Q  = (const float*)d_in[1];
    const float* LS = (const float*)d_in[2];
    const float* CB = (const float*)d_in[3];

    const int M  = in_sizes[0];                        // 131072 rows
    float* emb   = (float*)d_out;                      // [M, 256]
    float* probs = (float*)d_out + (size_t)M * EMBED;  // [M, 256]

    const size_t FRAG_OFF = 4096;
    const size_t WS_NEED  = FRAG_OFF + 128u * 1024u;

    dim3 grid(M / ROWS_PER_BLOCK, EMBED / COLS_PER_BLOCK); // (1024, 2)

    if (ws_size >= WS_NEED) {
        float*    negS = (float*)d_ws;
        _Float16* frag = (_Float16*)((char*)d_ws + FRAG_OFF);
        softq_prep<<<32, 256, 0, stream>>>(LS, CB, negS, frag);
        softq_main<<<grid, 256, 0, stream>>>(X, Q, negS, frag, emb, probs);
    } else {
        softq_fused<<<grid, 256, 0, stream>>>(X, Q, LS, CB, emb, probs);
    }
}
